// GCF_76587856822392
// MI455X (gfx1250) — matrix operand order repaired and checked by reading, not compiled, not run on hardware
//
#include <hip/hip_runtime.h>
#include <math.h>

#define BB 64
#define LL 16
#define EE 50000
#define RR 512
#define DWD 300
#define DWP 320      // DW padded to multiple of 32
#define DHD 768
#define HHD 384
#define G3H 1152
#define NSTEPS 3

typedef __attribute__((ext_vector_type(16))) _Float16 v16h;
typedef __attribute__((ext_vector_type(8)))  _Float16 v8h;
typedef __attribute__((ext_vector_type(8)))  float    v8f;

__device__ __forceinline__ float sigm(float x) { return 1.f / (1.f + __expf(-x)); }

// A-matrix 16x32 f16 fragment: lane half hi=0 -> K 0..15, hi=1 -> K 16..31 (16 consecutive K)
__device__ __forceinline__ v16h fragA_h(const _Float16* row, int k0, int khalf) {
  v8h lo = *(const v8h*)(row + k0 + khalf);
  v8h hi = *(const v8h*)(row + k0 + khalf + 8);
  return __builtin_shufflevector(lo, hi, 0, 1, 2, 3, 4, 5, 6, 7, 8, 9, 10, 11, 12, 13, 14, 15);
}

// B-matrix 32x16 f16 fragment: lanes 0-15 hold K=0..15, lanes 16-31 hold K=16..31 (16 consecutive K)
__device__ __forceinline__ v16h fragB_h(const _Float16* row, int k0, int kbase) {
  v8h lo = *(const v8h*)(row + k0 + kbase);
  v8h hi = *(const v8h*)(row + k0 + kbase + 8);
  return __builtin_shufflevector(lo, hi, 0, 1, 2, 3, 4, 5, 6, 7, 8, 9, 10, 11, 12, 13, 14, 15);
}

// C[M,N] = act(A[M,K] @ W[N,K]^T + bias[N]); M mult of 16, N mult of 32, K mult of 32.
// 256 thr = 8 waves; each wave computes a 16x32 tile (two adjacent N-tiles sharing fragA).
__global__ void k_gemm_nt(const _Float16* __restrict__ A, const _Float16* __restrict__ W,
                          const float* __restrict__ bias, float* __restrict__ C,
                          int M, int N, int K, int act) {
  const int wave = threadIdx.x >> 5;
  const int lane = threadIdx.x & 31;
  const int m16 = lane & 15;
  const int hi  = lane >> 4;
  const int tiles_n2 = N >> 5;
  const int tiles_m  = M >> 4;
  int tile = blockIdx.x * 8 + wave;
  if (tile >= tiles_m * tiles_n2) return;
  const int tm = tile / tiles_n2;
  const int tn = tile % tiles_n2;
  const _Float16* arow  = A + (size_t)(tm * 16 + m16) * K;
  const _Float16* wrow0 = W + (size_t)(tn * 32 + m16) * K;
  const _Float16* wrow1 = W + (size_t)(tn * 32 + 16 + m16) * K;
  v8f acc0 = {}, acc1 = {};
  for (int k0 = 0; k0 < K; k0 += 32) {
    __builtin_prefetch((const void*)(wrow0 + k0 + 128), 0, 0);
    __builtin_prefetch((const void*)(wrow1 + k0 + 128), 0, 0);
    v16h a  = fragA_h(arow, k0, hi * 16);
    v16h b0 = fragB_h(wrow0, k0, hi * 16);
    v16h b1 = fragB_h(wrow1, k0, hi * 16);
    acc0 = __builtin_amdgcn_wmma_f32_16x16x32_f16(false, a, false, b0, (short)0, acc0, false, false);
    acc1 = __builtin_amdgcn_wmma_f32_16x16x32_f16(false, a, false, b1, (short)0, acc1, false, false);
  }
  const int n0 = tn * 32 + m16;
  const int n1 = n0 + 16;
  const float bv0 = bias ? bias[n0] : 0.f;
  const float bv1 = bias ? bias[n1] : 0.f;
#pragma unroll
  for (int r = 0; r < 8; ++r) {
    int m = tm * 16 + r + 8 * hi;
    float v0 = acc0[r] + bv0;
    float v1 = acc1[r] + bv1;
    if (act == 1) { v0 = sigm(v0); v1 = sigm(v1); }
    C[(size_t)m * N + n0] = v0;
    C[(size_t)m * N + n1] = v1;
  }
}

// f32 -> f16 conversion (flat)
__global__ void k_cvt(const float* __restrict__ src, _Float16* __restrict__ dst, int n) {
  int idx = blockIdx.x * blockDim.x + threadIdx.x;
  if (idx < n) dst[idx] = (_Float16)src[idx];
}

// f32 [rows,K] -> f16 [rows,KP] zero-padded
__global__ void k_cvt_pad(const float* __restrict__ src, _Float16* __restrict__ dst,
                          int rows, int K, int KP) {
  int idx = blockIdx.x * blockDim.x + threadIdx.x;
  if (idx >= rows * KP) return;
  int r = idx / KP, k = idx % KP;
  dst[idx] = (k < K) ? (_Float16)src[(size_t)r * K + k] : (_Float16)0.f;
}

// embedding gather -> f16 padded [B*L, DWP]
__global__ void k_embed(const int* __restrict__ q, const float* __restrict__ wemb,
                        _Float16* __restrict__ qemb) {
  int idx = blockIdx.x * blockDim.x + threadIdx.x;
  if (idx >= BB * LL * DWP) return;
  int k = idx % DWP;
  int bl = idx / DWP;
  qemb[idx] = (k < DWD) ? (_Float16)wemb[(size_t)q[bl] * DWD + k] : (_Float16)0.f;
}

__global__ void k_mask(const int* __restrict__ q, float* __restrict__ amask) {
  int b = threadIdx.x;
  if (b >= BB) return;
  int cnt = 0;
  for (int l = 0; l < LL; ++l) cnt += (q[b * LL + l] == 0) ? 1 : 0;
  int qlen = LL - cnt;
  for (int l = 0; l < LL; ++l) amask[b * LL + l] = (l < qlen) ? 1.f : 0.f;
}

// Fused GRU: 16 batch rows per block (grid=4), h in LDS (f32 + f16 shadow),
// gh = h @ Whh^T via WMMA (72 N-tiles / 8 waves), cell fused.
__global__ void k_gru(const float* __restrict__ xp, const _Float16* __restrict__ WhhH,
                      const float* __restrict__ bhh, float* __restrict__ hs, int reverse) {
  __shared__ float h_s[16 * HHD];
  __shared__ alignas(16) _Float16 h_h[16 * HHD];
  __shared__ float gh_s[16 * G3H];
  const int tid = threadIdx.x;
  const int wave = tid >> 5;
  const int lane = tid & 31;
  const int b0 = blockIdx.x * 16;
  const int m16 = lane & 15;
  const int hi = lane >> 4;
  for (int e = tid; e < 16 * HHD; e += 256) { h_s[e] = 0.f; h_h[e] = (_Float16)0.f; }
  __syncthreads();
  for (int step = 0; step < LL; ++step) {
    int l = reverse ? (LL - 1 - step) : step;
    for (int q = 0; q < 9; ++q) {
      int n0 = (wave * 9 + q) * 16;
      const _Float16* arow = h_h + m16 * HHD;
      const _Float16* wrow = WhhH + (size_t)(n0 + m16) * HHD;
      v8f acc = {};
      for (int k0 = 0; k0 < HHD; k0 += 32) {
        v16h a  = fragA_h(arow, k0, hi * 16);
        v16h bf = fragB_h(wrow, k0, hi * 16);
        acc = __builtin_amdgcn_wmma_f32_16x16x32_f16(false, a, false, bf, (short)0, acc, false, false);
      }
      int n = n0 + m16;
#pragma unroll
      for (int r = 0; r < 8; ++r) gh_s[(r + 8 * hi) * G3H + n] = acc[r];
    }
    __syncthreads();
    for (int e = tid; e < 16 * HHD; e += 256) {
      int mb = e / HHD, hh = e % HHD;
      int b = b0 + mb;
      size_t xpo = ((size_t)b * LL + l) * G3H;
      float hr = gh_s[mb * G3H + hh]           + bhh[hh];
      float hz = gh_s[mb * G3H + HHD + hh]     + bhh[HHD + hh];
      float hn = gh_s[mb * G3H + 2 * HHD + hh] + bhh[2 * HHD + hh];
      float xr = xp[xpo + hh], xz = xp[xpo + HHD + hh], xn = xp[xpo + 2 * HHD + hh];
      float hp = h_s[e];
      float rg = sigm(xr + hr);
      float zg = sigm(xz + hz);
      float ng = tanhf(xn + rg * hn);
      float hnw = (1.f - zg) * ng + zg * hp;
      h_s[e] = hnw;
      h_h[e] = (_Float16)hnw;
      hs[((size_t)b * LL + l) * HHD + hh] = hnw;
    }
    __syncthreads();
  }
}

__global__ void k_concat(const float* __restrict__ hf, const float* __restrict__ hb,
                         float* __restrict__ qwh, float* __restrict__ hopbuf,
                         float* __restrict__ dist) {
  int idx = blockIdx.x * blockDim.x + threadIdx.x;
  if (idx >= BB * LL * DHD) return;
  int d = idx % DHD;
  int bl = idx / DHD;
  float v = (d < HHD) ? hf[(size_t)bl * HHD + d] : hb[(size_t)bl * HHD + (d - HHD)];
  qwh[idx] = v; hopbuf[idx] = v; dist[idx] = 0.f;
}

// Per-batch: logits S[m][l] = h_key[b,l]·qwh[b,m]; masked row softmax; col-attn; hop_ctx
__global__ void k_attn(const float* __restrict__ hkey, const float* __restrict__ qwh,
                       const float* __restrict__ hop, const float* __restrict__ amask,
                       float* __restrict__ hopctx, float* __restrict__ attbuf) {
  const int b = blockIdx.x;
  const int tid = threadIdx.x;
  __shared__ float S[16][16];
  __shared__ float att[16];
  {
    int i = tid >> 4, j = tid & 15;
    const float* hk = hkey + ((size_t)b * LL + j) * DHD;
    const float* qh = qwh + ((size_t)b * LL + i) * DHD;
    float acc = 0.f;
    for (int d = 0; d < DHD; ++d) acc += hk[d] * qh[d];
    S[i][j] = acc;
  }
  __syncthreads();
  if (tid < 16) {
    int i = tid;
    float mx = -1e30f;
    for (int j = 0; j < 16; ++j) mx = fmaxf(mx, S[i][j]);
    float p[16], sum = 0.f;
    for (int j = 0; j < 16; ++j) { p[j] = __expf(S[i][j] - mx); sum += p[j]; }
    float s2 = 0.f;
    for (int j = 0; j < 16; ++j) { p[j] = p[j] / sum * amask[b * LL + j]; s2 += p[j]; }
    float inv = 1.f / (s2 + 1e-6f);
    for (int j = 0; j < 16; ++j) S[i][j] = p[j] * inv;
  }
  __syncthreads();
  if (tid < 16) {
    int j = tid; float cs = 0.f;
    for (int i = 0; i < 16; ++i) cs += S[i][j];
    att[j] = cs;
  }
  __syncthreads();
  if (tid == 0) {
    float mx = -1e30f;
    for (int j = 0; j < 16; ++j) mx = fmaxf(mx, att[j]);
    float sum = 0.f;
    for (int j = 0; j < 16; ++j) { att[j] = __expf(att[j] - mx); sum += att[j]; }
    float s2 = 0.f;
    for (int j = 0; j < 16; ++j) { att[j] /= sum; s2 += att[j]; }
    float inv2 = 1.f / (s2 + 1e-6f);
    for (int j = 0; j < 16; ++j) att[j] *= inv2;
  }
  __syncthreads();
  for (int e = tid; e < LL * DHD; e += 256) {
    int i = e / DHD, d = e % DHD;
    float s = 0.f;
#pragma unroll
    for (int j = 0; j < 16; ++j) s += S[i][j] * hop[((size_t)b * LL + j) * DHD + d];
    hopctx[((size_t)b * LL + i) * DHD + d] = s;
  }
  if (tid < 16) attbuf[b * LL + tid] = att[tid];
}

__global__ void k_update(const float* __restrict__ qwh, const float* __restrict__ hopctx,
                         const float* __restrict__ zbuf, float* __restrict__ hopbuf,
                         float* __restrict__ dist, int t, int n) {
  int idx = blockIdx.x * blockDim.x + threadIdx.x;
  if (idx >= n) return;
  float hc = hopctx[idx];
  if (t == 0) { hopbuf[idx] = qwh[idx] + hc; dist[idx] = hc; }
  else {
    float dn = hc + zbuf[idx] * dist[idx];
    hopbuf[idx] = qwh[idx] + dn;
    dist[idx] = dn;
  }
}

__global__ void k_ctxh(const float* __restrict__ hopbuf, const float* __restrict__ attbuf,
                       float* __restrict__ ctxs, float* __restrict__ ctxh, int t) {
  int idx = blockIdx.x * blockDim.x + threadIdx.x;
  if (idx >= BB * DHD) return;
  int b = idx / DHD, d = idx % DHD;
  float s = 0.f;
  for (int j = 0; j < LL; ++j) s += hopbuf[((size_t)b * LL + j) * DHD + d] * attbuf[b * LL + j];
  ctxs[idx] = s;
  ctxh[((size_t)b * NSTEPS + t) * DHD + d] = s;
}

__global__ void k_zero(float* __restrict__ x, int n) {
  int idx = blockIdx.x * blockDim.x + threadIdx.x;
  if (idx < n) x[idx] = 0.f;
}

__global__ void k_follow(const float* __restrict__ eprev, const float* __restrict__ rel,
                         const int* __restrict__ subj, const int* __restrict__ ridx,
                         const int* __restrict__ obj, float* __restrict__ eout, int T) {
  long long idx = (long long)blockIdx.x * blockDim.x + threadIdx.x;
  if (idx >= (long long)T * BB) return;
  int t = (int)(idx >> 6);
  int b = (int)(idx & 63);
  float ev = eprev[(size_t)b * EE + subj[t]];
  if (ev == 0.f) return;
  float v = ev * rel[b * RR + ridx[t]];
  if (v != 0.f) atomicAdd(&eout[(size_t)b * EE + obj[t]], v);
}

__global__ void k_clamp(float* __restrict__ e, int n) {
  int idx = blockIdx.x * blockDim.x + threadIdx.x;
  if (idx >= n) return;
  float v = e[idx];
  if (v > 1.f) e[idx] = 1.f;   // e / (m*e + (1-m)) == 1 when e > 1
}

__global__ void k_relargmax(const float* __restrict__ relc, const float* __restrict__ relp,
                            float* __restrict__ m2) {
  int b = threadIdx.x;
  if (b >= BB) return;
  int ac = 0; float bc = relc[(size_t)b * RR];
  int ap = 0; float bp = relp[(size_t)b * RR];
  for (int k = 1; k < RR; ++k) {
    float vc = relc[(size_t)b * RR + k]; if (vc > bc) { bc = vc; ac = k; }
    float vp = relp[(size_t)b * RR + k]; if (vp > bp) { bp = vp; ap = k; }
  }
  int d = ap - ac; if (d < 0) d = -d;
  int mn = (ap < ac) ? ap : ac;
  m2[b] = (d == 1 && (mn % 2 == 0)) ? 1.f : 0.f;
}

__global__ void k_entmask(float* __restrict__ ent, const float* __restrict__ pp,
                          const float* __restrict__ m2) {
  int idx = blockIdx.x * blockDim.x + threadIdx.x;
  if (idx >= BB * EE) return;
  int b = idx / EE;
  ent[idx] *= (1.f - m2[b] * ((pp[idx] > 0.9f) ? 1.f : 0.f));
}

__global__ void k_hopattn(const float* __restrict__ ctxh, const float* __restrict__ hopW,
                          const float* __restrict__ hopb, float* __restrict__ hattn,
                          float* __restrict__ locf) {
  int b = threadIdx.x;
  if (b >= BB) return;
  float lg[NSTEPS];
  for (int s = 0; s < NSTEPS; ++s) {
    const float* c = ctxh + ((size_t)b * NSTEPS + s) * DHD;
    float acc = hopb[0];
    for (int d = 0; d < DHD; ++d) acc += c[d] * hopW[d];
    lg[s] = acc;
  }
  float mx = fmaxf(lg[0], fmaxf(lg[1], lg[2]));
  float e0 = __expf(lg[0] - mx), e1 = __expf(lg[1] - mx), e2 = __expf(lg[2] - mx);
  float inv = 1.f / (e0 + e1 + e2);
  float a0 = e0 * inv, a1 = e1 * inv, a2 = e2 * inv;
  hattn[b * 3 + 0] = a0; hattn[b * 3 + 1] = a1; hattn[b * 3 + 2] = a2;
  int loc = 0; float bv = a0;
  if (a1 > bv) { bv = a1; loc = 1; }
  if (a2 > bv) { bv = a2; loc = 2; }
  locf[b] = (loc > 1) ? 1.f : 0.f;
}

__global__ void k_final(const float* __restrict__ e0, const float* __restrict__ e1,
                        const float* __restrict__ e2, const float* __restrict__ hattn,
                        const float* __restrict__ locf, float* __restrict__ out) {
  int idx = blockIdx.x * blockDim.x + threadIdx.x;
  if (idx >= BB * EE) return;
  int b = idx / EE;
  float a0 = hattn[b * 3], a1 = hattn[b * 3 + 1], a2 = hattn[b * 3 + 2];
  float h0 = e0[idx];
  float m1 = 1.f - ((h0 > 0.f) ? 1.f : 0.f) * locf[b];
  float disc = m1 * 0.1f + 0.9f;
  out[idx] = (h0 * a0 + e1[idx] * a1 + e2[idx] * a2) * disc;
}

extern "C" void kernel_launch(void* const* d_in, const int* in_sizes, int n_in,
                              void* d_out, int out_size, void* d_ws, size_t ws_size,
                              hipStream_t stream) {
  const int*   questions = (const int*)d_in[0];
  const float* e_s   = (const float*)d_in[1];
  const float* wemb  = (const float*)d_in[2];
  const float* gfWih = (const float*)d_in[3];
  const float* gfWhh = (const float*)d_in[4];
  const float* gfbih = (const float*)d_in[5];
  const float* gfbhh = (const float*)d_in[6];
  const float* gbWih = (const float*)d_in[7];
  const float* gbWhh = (const float*)d_in[8];
  const float* gbbih = (const float*)d_in[9];
  const float* gbbhh = (const float*)d_in[10];
  const float* stepW = (const float*)d_in[11];
  const float* stepb = (const float*)d_in[12];
  const float* relW  = (const float*)d_in[13];
  const float* relb  = (const float*)d_in[14];
  const float* highW = (const float*)d_in[15];
  const float* highb = (const float*)d_in[16];
  const float* hopW  = (const float*)d_in[17];
  const float* hopb  = (const float*)d_in[18];
  const int* subj = (const int*)d_in[19];
  const int* ridx = (const int*)d_in[20];
  const int* obj  = (const int*)d_in[21];
  const int T = in_sizes[19];

  char* p = (char*)d_ws;
  auto allocF = [&](size_t nf) -> float* {
    float* r = (float*)p;
    p += ((nf * sizeof(float)) + 255) & ~(size_t)255;
    return r;
  };
  auto allocH = [&](size_t nh) -> _Float16* {
    _Float16* r = (_Float16*)p;
    p += ((nh * sizeof(_Float16)) + 255) & ~(size_t)255;
    return r;
  };
  // f32 activations
  float* amask  = allocF(BB * LL);
  float* xp_f   = allocF((size_t)BB * LL * G3H);
  float* xp_b   = allocF((size_t)BB * LL * G3H);
  float* hs_f   = allocF((size_t)BB * LL * HHD);
  float* hs_b   = allocF((size_t)BB * LL * HHD);
  float* qwh    = allocF((size_t)BB * LL * DHD);
  float* hopbuf = allocF((size_t)BB * LL * DHD);
  float* dist   = allocF((size_t)BB * LL * DHD);
  float* hkey   = allocF((size_t)BB * LL * DHD);
  float* zbuf   = allocF((size_t)BB * LL * DHD);
  float* hopctx = allocF((size_t)BB * LL * DHD);
  float* attbuf = allocF(BB * LL);
  float* ctxs   = allocF((size_t)BB * DHD);
  float* ctxh   = allocF((size_t)BB * NSTEPS * DHD);
  float* relp   = allocF((size_t)NSTEPS * BB * RR);
  float* ent    = allocF((size_t)NSTEPS * BB * EE);
  float* m2     = allocF(BB);
  float* hattn  = allocF(BB * NSTEPS);
  float* locf   = allocF(BB);
  // f16 operands
  _Float16* qembH  = allocH((size_t)BB * LL * DWP);
  _Float16* wihfH  = allocH((size_t)G3H * DWP);
  _Float16* wihbH  = allocH((size_t)G3H * DWP);
  _Float16* whhfH  = allocH((size_t)G3H * HHD);
  _Float16* whhbH  = allocH((size_t)G3H * HHD);
  _Float16* stepWH = allocH((size_t)NSTEPS * DHD * DHD);
  _Float16* highWH = allocH((size_t)DHD * DHD);
  _Float16* relWH  = allocH((size_t)RR * DHD);
  _Float16* hopH   = allocH((size_t)BB * LL * DHD);
  _Float16* distH  = allocH((size_t)BB * LL * DHD);
  _Float16* ctxsH  = allocH((size_t)BB * DHD);

  auto gemm = [&](const _Float16* A, const _Float16* W, const float* bias, float* C,
                  int M, int N, int K, int act) {
    int tiles = (M >> 4) * (N >> 5);
    int blocks = (tiles + 7) >> 3;
    k_gemm_nt<<<blocks, 256, 0, stream>>>(A, W, bias, C, M, N, K, act);
  };
  auto cvt = [&](const float* s, _Float16* d, int n) {
    k_cvt<<<(n + 255) / 256, 256, 0, stream>>>(s, d, n);
  };

  // one-time weight conversions
  { int n = G3H * DWP; k_cvt_pad<<<(n + 255) / 256, 256, 0, stream>>>(gfWih, wihfH, G3H, DWD, DWP); }
  { int n = G3H * DWP; k_cvt_pad<<<(n + 255) / 256, 256, 0, stream>>>(gbWih, wihbH, G3H, DWD, DWP); }
  cvt(gfWhh, whhfH, G3H * HHD);
  cvt(gbWhh, whhbH, G3H * HHD);
  cvt(stepW, stepWH, NSTEPS * DHD * DHD);
  cvt(highW, highWH, DHD * DHD);
  cvt(relW, relWH, RR * DHD);

  { int n = BB * LL * DWP; k_embed<<<(n + 255) / 256, 256, 0, stream>>>(questions, wemb, qembH); }
  k_mask<<<1, 64, 0, stream>>>(questions, amask);

  gemm(qembH, wihfH, gfbih, xp_f, BB * LL, G3H, DWP, 0);
  gemm(qembH, wihbH, gbbih, xp_b, BB * LL, G3H, DWP, 0);
  k_gru<<<4, 256, 0, stream>>>(xp_f, whhfH, gfbhh, hs_f, 0);
  k_gru<<<4, 256, 0, stream>>>(xp_b, whhbH, gbbhh, hs_b, 1);
  { int n = BB * LL * DHD; k_concat<<<(n + 255) / 256, 256, 0, stream>>>(hs_f, hs_b, qwh, hopbuf, dist); }

  for (int t = 0; t < NSTEPS; ++t) {
    cvt(hopbuf, hopH, BB * LL * DHD);
    gemm(hopH, stepWH + (size_t)t * DHD * DHD, stepb + t * DHD, hkey, BB * LL, DHD, DHD, 0);
    k_attn<<<BB, 256, 0, stream>>>(hkey, qwh, hopbuf, amask, hopctx, attbuf);
    if (t > 0) {
      cvt(dist, distH, BB * LL * DHD);
      gemm(distH, highWH, highb, zbuf, BB * LL, DHD, DHD, 1);
    }
    { int n = BB * LL * DHD; k_update<<<(n + 255) / 256, 256, 0, stream>>>(qwh, hopctx, zbuf, hopbuf, dist, t, n); }
    { int n = BB * DHD; k_ctxh<<<(n + 255) / 256, 256, 0, stream>>>(hopbuf, attbuf, ctxs, ctxh, t); }

    float* rt = relp + (size_t)t * BB * RR;
    cvt(ctxs, ctxsH, BB * DHD);
    gemm(ctxsH, relWH, relb, rt, BB, RR, DHD, 1);

    float* et = ent + (size_t)t * BB * EE;
    { int n = BB * EE; k_zero<<<(n + 255) / 256, 256, 0, stream>>>(et, n); }
    const float* ep = (t == 0) ? e_s : (ent + (size_t)(t - 1) * BB * EE);
    { long long n = (long long)T * BB; int blocks = (int)((n + 255) / 256);
      k_follow<<<blocks, 256, 0, stream>>>(ep, rt, subj, ridx, obj, et, T); }
    { int n = BB * EE; k_clamp<<<(n + 255) / 256, 256, 0, stream>>>(et, n); }

    if (t > 0) {
      k_relargmax<<<1, 64, 0, stream>>>(rt, relp + (size_t)(t - 1) * BB * RR, m2);
      const float* pp = (t == 1) ? e_s : ent;  // ent_probs[0] for t==2
      { int n = BB * EE; k_entmask<<<(n + 255) / 256, 256, 0, stream>>>(et, pp, m2); }
    }
  }

  k_hopattn<<<1, 64, 0, stream>>>(ctxh, hopW, hopb, hattn, locf);
  { int n = BB * EE;
    k_final<<<(n + 255) / 256, 256, 0, stream>>>(ent, ent + (size_t)BB * EE,
                                                 ent + (size_t)2 * BB * EE, hattn, locf,
                                                 (float*)d_out); }
}
